// MambaBlock_55198919688641
// MI455X (gfx1250) — compile-verified, matrix operand order checked by reading, not run on hardware
//
#include <hip/hip_runtime.h>
#include <hip/hip_bf16.h>
#include <math.h>

// ---- problem constants (match reference) ----
#define D_MODEL 1024
#define D_STATE 16
#define D_CONV  4
#define D_INNER 2048
#define DT_RANK 64
#define BB 2
#define TT 1024
#define ROWS (BB*TT)                  // 2048 token rows
#define XDBC_W (DT_RANK + 2*D_STATE)  // 96

typedef __attribute__((ext_vector_type(16))) _Float16 v16h;
typedef __attribute__((ext_vector_type(8)))  _Float16 v8h;
typedef __attribute__((ext_vector_type(8)))  float    v8f;

// global (address-space 1) pointer types: force global_load_b128 (not flat)
typedef const __attribute__((address_space(1))) _Float16* gptr_t;
typedef const __attribute__((address_space(1))) v8h*      gv8p_t;

__device__ __forceinline__ float sigmoidf_(float v){ return 1.0f/(1.0f+__expf(-v)); }

// ---------------- f32 -> f16 conversion (weights) ----------------
__global__ void k_f32_to_f16(const float* __restrict__ src, _Float16* __restrict__ dst, int n){
  int i = blockIdx.x*blockDim.x + threadIdx.x;
  if (i < n) dst[i] = (_Float16)src[i];
}

// ---------------- LayerNorm: one block per token row ----------------
__global__ void k_layernorm(const float* __restrict__ x, const float* __restrict__ gamma,
                            const float* __restrict__ beta, _Float16* __restrict__ xn){
  __shared__ float s_sum[256], s_sq[256];
  const int row = blockIdx.x;
  const float* xr = x + (size_t)row*D_MODEL;
  float s = 0.f, q = 0.f;
  for (int c = threadIdx.x; c < D_MODEL; c += blockDim.x){ float v = xr[c]; s += v; q += v*v; }
  s_sum[threadIdx.x] = s; s_sq[threadIdx.x] = q; __syncthreads();
  for (int off = 128; off > 0; off >>= 1){
    if ((int)threadIdx.x < off){ s_sum[threadIdx.x] += s_sum[threadIdx.x+off];
                                 s_sq[threadIdx.x]  += s_sq[threadIdx.x+off]; }
    __syncthreads();
  }
  const float mu   = s_sum[0] * (1.0f/D_MODEL);
  const float var  = s_sq[0] * (1.0f/D_MODEL) - mu*mu;
  const float rstd = rsqrtf(var + 1e-5f);
  _Float16* o = xn + (size_t)row*D_MODEL;
  for (int c = threadIdx.x; c < D_MODEL; c += blockDim.x){
    float v = (xr[c]-mu)*rstd*gamma[c] + beta[c];
    o[c] = (_Float16)v;
  }
}

// ---------------- WMMA GEMM:  C[M,N] = A[M,K] * B[N,K]^T  (+ fused epilogue) ----
// A, B: f16 row-major, K contiguous; C: f32 row-major (ldc = N).
// One wave computes a (16*MT) x (16*NT) block; K double-buffered in 2x32 strips
// (requires K % 64 == 0 — true for all call sites: 1024, 2048, 64).
// Row pointers live in address space 1 and advance by 64 elements per step;
// intra-strip offsets are immediates folded into global_load_b128.
// EPI = 0: plain store; 1: softplus(acc + E[n]); 2: acc + E[m*N+n] (residual).
template<int MT, int NT, int EPI>
__global__ __launch_bounds__(256, 1)
void k_gemm_f16(const _Float16* __restrict__ A, const _Float16* __restrict__ Bm,
                float* __restrict__ C, const float* __restrict__ E,
                int M, int N, int K){
  const int lane  = threadIdx.x & 31;
  const int gwave = (int)((blockIdx.x*blockDim.x + threadIdx.x) >> 5);
  const int groupsN = N / (16*NT);
  const int strips  = (M / (16*MT)) * groupsN;
  if (gwave >= strips) return;                // wave-uniform: EXEC stays all-ones
  const int m0 = (gwave / groupsN) * (16*MT);
  const int n0 = (gwave % groupsN) * (16*NT);
  const int r  = lane & 15;
  const int hi = (lane >> 4) << 3;            // 0 or 8

  // live row pointers (global AS), pre-biased by the lane's K sub-chunk
  gptr_t pa[MT];
  gptr_t pb[NT];
  #pragma unroll
  for (int mi = 0; mi < MT; ++mi) pa[mi] = (gptr_t)(A  + (size_t)(m0 + 16*mi + r) * K + hi);
  #pragma unroll
  for (int j  = 0; j  < NT; ++j ) pb[j]  = (gptr_t)(Bm + (size_t)(n0 + 16*j  + r) * K + hi);

  auto ld = [&](gptr_t p, int off) -> v16h {
    v8h lo = *(gv8p_t)(p + off);              // imm offsets: 0/32 (+16 for upper)
    v8h up = *(gv8p_t)(p + off + 16);
    return __builtin_shufflevector(lo, up, 0,1,2,3,4,5,6,7,8,9,10,11,12,13,14,15);
  };

  v8f acc[MT][NT];
  #pragma unroll
  for (int mi = 0; mi < MT; ++mi)
    #pragma unroll
    for (int j = 0; j < NT; ++j) acc[mi][j] = (v8f){};

  // ---- ping-pong double buffer over 32-wide K strips ----
  v16h a0[MT], b0[NT], a1[MT], b1[NT];
  #pragma unroll
  for (int mi = 0; mi < MT; ++mi) a0[mi] = ld(pa[mi], 0);
  #pragma unroll
  for (int j = 0; j < NT; ++j)    b0[j]  = ld(pb[j], 0);

  const int nloops = K >> 6;                  // K/64
  for (int it = 0; it < nloops; ++it){
    // refill buffer 1 (strip +32) while buffer 0 computes
    #pragma unroll
    for (int mi = 0; mi < MT; ++mi) a1[mi] = ld(pa[mi], 32);
    #pragma unroll
    for (int j = 0; j < NT; ++j)    b1[j]  = ld(pb[j], 32);
    if (it + 8 < nloops){                     // L2 stream-ahead prefetch
      __builtin_prefetch((const _Float16*)(pa[0] + 512), 0, 3);
      __builtin_prefetch((const _Float16*)(pb[0] + 512), 0, 3);
    }
    #pragma unroll
    for (int mi = 0; mi < MT; ++mi)
      #pragma unroll
      for (int j = 0; j < NT; ++j)
        acc[mi][j] = __builtin_amdgcn_wmma_f32_16x16x32_f16(
            false, a0[mi], false, b0[j], (short)0, acc[mi][j], false, false);
    // advance pointers, refill buffer 0 (next strip) while buffer 1 computes
    #pragma unroll
    for (int mi = 0; mi < MT; ++mi) pa[mi] += 64;
    #pragma unroll
    for (int j = 0; j < NT; ++j)    pb[j]  += 64;
    if (it + 1 < nloops){
      #pragma unroll
      for (int mi = 0; mi < MT; ++mi) a0[mi] = ld(pa[mi], 0);
      #pragma unroll
      for (int j = 0; j < NT; ++j)    b0[j]  = ld(pb[j], 0);
    }
    #pragma unroll
    for (int mi = 0; mi < MT; ++mi)
      #pragma unroll
      for (int j = 0; j < NT; ++j)
        acc[mi][j] = __builtin_amdgcn_wmma_f32_16x16x32_f16(
            false, a1[mi], false, b1[j], (short)0, acc[mi][j], false, false);
  }

  // D layout: lane -> N = base + (lane&15); VGPR v -> M = base + v + 8*(lane>=16)
  #pragma unroll
  for (int mi = 0; mi < MT; ++mi){
    const int mb = m0 + 16*mi + ((lane >> 4) << 3);
    #pragma unroll
    for (int j = 0; j < NT; ++j){
      const int n = n0 + 16*j + r;
      #pragma unroll
      for (int v = 0; v < 8; ++v){
        float val = acc[mi][j][v];
        if (EPI == 1){ val += E[n]; val = (val > 20.f) ? val : log1pf(__expf(val)); }
        if (EPI == 2){ val += E[(size_t)(mb + v) * N + n]; }
        C[(size_t)(mb + v) * N + n] = val;
      }
    }
  }
}

// ---------------- causal depthwise conv1d (k=4) + SiLU ----------------
// xz: [ROWS, 2*D_INNER] f32 (xb = cols [0,2048), z = cols [2048,4096))
__global__ void k_conv_silu(const float* __restrict__ xz, const float* __restrict__ Wc,
                            const float* __restrict__ bc, float* __restrict__ xb32,
                            _Float16* __restrict__ xb16){
  int i = blockIdx.x*blockDim.x + threadIdx.x;        // over ROWS * D_INNER
  if (i >= ROWS*D_INNER) return;
  const int d   = i % D_INNER;
  const int row = i / D_INNER;                        // b*T + t
  const int t   = row % TT;
  const int b   = row / TT;
  float acc = bc[d];
  #pragma unroll
  for (int j = 0; j < D_CONV; ++j){
    int tp = t - (D_CONV-1) + j;
    if (tp >= 0)
      acc += Wc[d*D_CONV + j] * xz[((size_t)(b*TT + tp))*(2*D_INNER) + d];
  }
  float v = acc * sigmoidf_(acc);
  xb32[i] = v;
  xb16[i] = (_Float16)v;
}

// ---------------- slice dt_r = x_dbc[:, 0:64] -> f16 ----------------
__global__ void k_slice_dtr(const float* __restrict__ xdbc, _Float16* __restrict__ dtr){
  int i = blockIdx.x*blockDim.x + threadIdx.x;        // ROWS * DT_RANK
  if (i >= ROWS*DT_RANK) return;
  int rrow = i / DT_RANK, c = i % DT_RANK;
  dtr[i] = (_Float16)xdbc[(size_t)rrow*XDBC_W + c];
}

// ---------------- selective scan + D-skip + SiLU(z) gate ----------------
// One thread per (b, d); h[16] in registers; serial over T.
// B/C rows are identical across a block (same b, same t sequence):
// stage 8 timesteps of B/C into LDS cooperatively, then broadcast via ds_load.
#define SCAN_CHUNK 8
__global__ void k_scan(const float* __restrict__ dt, const float* __restrict__ xb32,
                       const float* __restrict__ xdbc, const float* __restrict__ xz,
                       const float* __restrict__ A_log, const float* __restrict__ Dp,
                       _Float16* __restrict__ y16){
  __shared__ float sBC[SCAN_CHUNK][2*D_STATE];        // [t][B(16) | C(16)]
  const int gid = blockIdx.x*blockDim.x + threadIdx.x;   // BB * D_INNER = 4096
  const int d = gid % D_INNER;
  const int b = gid / D_INNER;                           // uniform per block
  float a[D_STATE], h[D_STATE];
  #pragma unroll
  for (int n = 0; n < D_STATE; ++n){ a[n] = -__expf(A_log[d*D_STATE + n]); h[n] = 0.f; }
  const float Dd = Dp[d];

  for (int t0 = 0; t0 < TT; t0 += SCAN_CHUNK){
    {
      const int tloc = threadIdx.x >> 5;           // 0..7
      const int col  = threadIdx.x & 31;           // 0..31
      const size_t srow = (size_t)(b*TT + t0 + tloc);
      sBC[tloc][col] = xdbc[srow*XDBC_W + DT_RANK + col];
    }
    __syncthreads();
    #pragma unroll
    for (int tt = 0; tt < SCAN_CHUNK; ++tt){
      const size_t row = (size_t)(b*TT + t0 + tt);
      const float dtv = dt  [row*D_INNER + d];
      const float xv  = xb32[row*D_INNER + d];
      float y = Dd * xv;
      const float dtx = dtv * xv;
      #pragma unroll
      for (int n = 0; n < D_STATE; ++n){
        float ab = __expf(dtv * a[n]);             // v_exp_f32
        h[n] = ab*h[n] + dtx * sBC[tt][n];
        y += h[n] * sBC[tt][D_STATE + n];
      }
      const float z = xz[row*(2*D_INNER) + D_INNER + d];
      y *= z * sigmoidf_(z);
      y16[row*D_INNER + d] = (_Float16)y;
    }
    __syncthreads();
  }
}

// =======================================================================
extern "C" void kernel_launch(void* const* d_in, const int* in_sizes, int n_in,
                              void* d_out, int out_size, void* d_ws, size_t ws_size,
                              hipStream_t stream){
  (void)in_sizes; (void)n_in; (void)out_size; (void)ws_size;
  const float* x     = (const float*)d_in[0];
  const float* W_in  = (const float*)d_in[1];
  const float* W_c   = (const float*)d_in[2];
  const float* b_c   = (const float*)d_in[3];
  const float* W_x   = (const float*)d_in[4];
  const float* W_dt  = (const float*)d_in[5];
  const float* b_dt  = (const float*)d_in[6];
  const float* A_log = (const float*)d_in[7];
  const float* Dp    = (const float*)d_in[8];
  const float* W_out = (const float*)d_in[9];
  const float* gamma = (const float*)d_in[10];
  const float* beta  = (const float*)d_in[11];
  float* out = (float*)d_out;

  // ---- workspace layout ----
  char* ws = (char*)d_ws;
  size_t off = 0;
  auto alloc = [&](size_t bytes)->char*{ char* p = ws + off; off += (bytes + 255) & ~(size_t)255; return p; };
  _Float16* w_in16  = (_Float16*)alloc((size_t)2*D_INNER*D_MODEL*2);   // 8 MB
  _Float16* w_x16   = (_Float16*)alloc((size_t)XDBC_W*D_INNER*2);      // 384 KB
  _Float16* w_dt16  = (_Float16*)alloc((size_t)D_INNER*DT_RANK*2);     // 256 KB
  _Float16* w_out16 = (_Float16*)alloc((size_t)D_MODEL*D_INNER*2);     // 4 MB
  _Float16* xn16    = (_Float16*)alloc((size_t)ROWS*D_MODEL*2);        // 4 MB
  float*    xz      = (float*)   alloc((size_t)ROWS*2*D_INNER*4);      // 32 MB
  float*    xb32    = (float*)   alloc((size_t)ROWS*D_INNER*4);        // 16 MB
  _Float16* xb16    = (_Float16*)alloc((size_t)ROWS*D_INNER*2);        // 8 MB
  float*    xdbc    = (float*)   alloc((size_t)ROWS*XDBC_W*4);         // 768 KB
  _Float16* dtr16   = (_Float16*)alloc((size_t)ROWS*DT_RANK*2);        // 256 KB
  float*    dt32    = (float*)   alloc((size_t)ROWS*D_INNER*4);        // 16 MB
  _Float16* y16     = (_Float16*)alloc((size_t)ROWS*D_INNER*2);        // 8 MB

  const int BT = 256;                 // 8 waves / block
  auto cdiv = [](int a, int b){ return (a + b - 1) / b; };

  // 1) weights -> f16
  k_f32_to_f16<<<cdiv(2*D_INNER*D_MODEL, BT), BT, 0, stream>>>(W_in,  w_in16,  2*D_INNER*D_MODEL);
  k_f32_to_f16<<<cdiv(XDBC_W*D_INNER,   BT), BT, 0, stream>>>(W_x,   w_x16,   XDBC_W*D_INNER);
  k_f32_to_f16<<<cdiv(D_INNER*DT_RANK,  BT), BT, 0, stream>>>(W_dt,  w_dt16,  D_INNER*DT_RANK);
  k_f32_to_f16<<<cdiv(D_MODEL*D_INNER,  BT), BT, 0, stream>>>(W_out, w_out16, D_MODEL*D_INNER);

  // 2) LayerNorm -> xn (f16)
  k_layernorm<<<ROWS, BT, 0, stream>>>(x, gamma, beta, xn16);

  // 3) in_proj: [2048,1024] x [4096,1024]^T -> xz [2048,4096]
  {
    int strips = (ROWS/32) * ((2*D_INNER)/64);       // 4096
    k_gemm_f16<2,4,0><<<cdiv(strips,8), BT, 0, stream>>>(xn16, w_in16, xz, nullptr,
                                                         ROWS, 2*D_INNER, D_MODEL);
  }

  // 4) depthwise causal conv + SiLU -> xb32 / xb16
  k_conv_silu<<<cdiv(ROWS*D_INNER, BT), BT, 0, stream>>>(xz, W_c, b_c, xb32, xb16);

  // 5) x_dbc: [2048,2048] x [96,2048]^T -> [2048,96]   (N=96 -> NT=2 strips)
  {
    int strips = (ROWS/32) * (XDBC_W/32);            // 192
    k_gemm_f16<2,2,0><<<cdiv(strips,8), BT, 0, stream>>>(xb16, w_x16, xdbc, nullptr,
                                                         ROWS, XDBC_W, D_INNER);
  }

  // 6) dt_r slice -> f16 ; dt GEMM [2048,64] x [2048,64]^T with fused softplus(+b_dt)
  k_slice_dtr<<<cdiv(ROWS*DT_RANK, BT), BT, 0, stream>>>(xdbc, dtr16);
  {
    int strips = (ROWS/32) * (D_INNER/64);           // 2048
    k_gemm_f16<2,4,1><<<cdiv(strips,8), BT, 0, stream>>>(dtr16, w_dt16, dt32, b_dt,
                                                         ROWS, D_INNER, DT_RANK);
  }

  // 7) selective scan + gate -> y16
  k_scan<<<cdiv(BB*D_INNER, BT), BT, 0, stream>>>(dt32, xb32, xdbc, xz, A_log, Dp, y16);

  // 8) out_proj: [2048,2048] x [1024,2048]^T -> d_out, residual fused in epilogue
  {
    int strips = (ROWS/32) * (D_MODEL/64);           // 1024
    k_gemm_f16<2,4,2><<<cdiv(strips,8), BT, 0, stream>>>(y16, w_out16, out, x,
                                                         ROWS, D_MODEL, D_INNER);
  }
}
